// LSTMStacked_88880053223652
// MI455X (gfx1250) — matrix operand order repaired and checked by reading, not compiled, not run on hardware
//
#include <hip/hip_runtime.h>
#include <hip/hip_bf16.h>

// ---------------------------------------------------------------------------
// Fused 2-layer LSTM (B=512,T=1024,F=32,H0=64,H1=32) + FC head for gfx1250.
// One workgroup = 16 batch rows, 4 waves (wave32). All weights resident in
// VGPRs in WMMA B-layout; gates via v_wmma_f32_16x16x32_bf16 (f32 accum).
// h state goes D-layout -> A-layout through double-buffered LDS tiles
// (2 barriers per timestep). Activations via native v_tanh_f32.
// ---------------------------------------------------------------------------

typedef __attribute__((ext_vector_type(16))) __bf16 v16bf;
typedef __attribute__((ext_vector_type(8)))  float  v8f;

#define Bsz 512
#define Tsz 1024
#define Fsz 32
#define H0  64
#define H1  32

static __device__ __forceinline__ __bf16 f2bf(float f) { return (__bf16)f; }

static __device__ __forceinline__ float fast_tanh(float x) {
#if __has_builtin(__builtin_amdgcn_tanhf)
  return __builtin_amdgcn_tanhf(x);
#elif __has_builtin(__builtin_amdgcn_tanh_f32)
  return __builtin_amdgcn_tanh_f32(x);
#else
  float xc = fminf(fmaxf(x, -9.0f), 9.0f);
  float e  = __builtin_amdgcn_exp2f(xc * 2.8853900817779268f);  // exp(2x)
  return 1.0f - 2.0f * __builtin_amdgcn_rcpf(e + 1.0f);
#endif
}
// sigmoid(x + b) = 0.5*tanh(0.5*x + 0.5*b) + 0.5   (hb = 0.5*b precomputed)
static __device__ __forceinline__ float sigm_b(float x, float hb) {
  return fmaf(0.5f, fast_tanh(fmaf(0.5f, x, hb)), 0.5f);
}

// Inverse of the 16-bit A-matrix K mapping (ISA 7.12.2):
//  lane-half hi, packed half j  ->  K = (j<8 ? j : j+8) + 8*hi
static __device__ __forceinline__ void invA(int k, int& hi, int& j) {
  if      (k <  8) { hi = 0; j = k;      }
  else if (k < 16) { hi = 1; j = k - 8;  }
  else if (k < 24) { hi = 0; j = k - 8;  }
  else             { hi = 1; j = k - 16; }
}

// Load one 32x16 (KxN) bf16 B-tile into WMMA B-layout registers.
// B[k][n] = W[rowb+n][colb+k]  (gates = h @ W^T); half j <-> K = (j<8 ? j : j+8) + 8*hi.
static __device__ __forceinline__ v16bf loadB(const float* __restrict__ W,
                                              int ldw, int rowb, int colb,
                                              int lane) {
  const int n = lane & 15, hi = lane >> 4;
  v16bf t;
#pragma unroll
  for (int j = 0; j < 16; ++j)
    t[j] = f2bf(W[(size_t)(rowb + n) * ldw + colb + (j < 8 ? j : j + 8) + hi * 8]);
  return t;
}

static __device__ __forceinline__ v8f wmma_bf16(v16bf a, v16bf b, v8f c) {
  return __builtin_amdgcn_wmma_f32_16x16x32_bf16(
      /*neg_a=*/false, a, /*neg_b=*/false, b,
      /*c_mod=*/(short)0, c, /*reuse_a=*/false, /*reuse_b=*/false);
}

__global__ __launch_bounds__(128, 1)
void lstm2_fused_kernel(const float* __restrict__ x,
                        const float* __restrict__ Wih0, const float* __restrict__ Whh0,
                        const float* __restrict__ bih0, const float* __restrict__ bhh0,
                        const float* __restrict__ Wih1, const float* __restrict__ Whh1,
                        const float* __restrict__ bih1, const float* __restrict__ bhh1,
                        const float* __restrict__ fcw,  const float* __restrict__ fcb,
                        float* __restrict__ dout) {
  const int tid  = threadIdx.x;
  const int lane = tid & 31;
  const int wv   = tid >> 5;          // wave 0..3
  const int n    = lane & 15;
  const int hi   = lane >> 4;
  const int rowbase = blockIdx.x * 16;

  // Double-buffered LDS tiles (A-layout bf16): [buf][...]
  __shared__ __align__(32) __bf16 xA [2][32 * 16];      // x_t tile
  __shared__ __align__(32) __bf16 h0A[2][2][32 * 16];   // h0, 2 K-blocks
  __shared__ __align__(32) __bf16 h1A[2][32 * 16];      // h1, 1 K-block
  __shared__ float h1plain[16 * 32];                    // final h1 for FC head

  // ---- zero-init recurrent LDS state (h(0) = 0) ----
  for (int i = tid; i < 2 * 2 * 32 * 16; i += 128) ((__bf16*)h0A)[i] = f2bf(0.0f);
  for (int i = tid; i < 2 * 32 * 16;     i += 128) ((__bf16*)h1A)[i] = f2bf(0.0f);

  // ---- load all weights into registers (WMMA B-layout), once ----
  v16bf Bh0[4][2], Bx0[4];
  float hb0[4];   // 0.5*bias for sigmoid gates (i,f,o); full bias for g (idx 2)
#pragma unroll
  for (int g = 0; g < 4; ++g) {
    const int rb = g * H0 + 16 * wv;
    Bx0[g] = loadB(Wih0, Fsz, rb, 0, lane);
#pragma unroll
    for (int kb = 0; kb < 2; ++kb) Bh0[g][kb] = loadB(Whh0, H0, rb, kb * 32, lane);
    float b = bih0[rb + n] + bhh0[rb + n];
    hb0[g] = (g == 2) ? b : 0.5f * b;
  }

  v16bf Bx1[4][2], Bh1[4];
  float hb1[4] = {0.f, 0.f, 0.f, 0.f};
  {
    v16bf z;
#pragma unroll
    for (int j = 0; j < 16; ++j) z[j] = f2bf(0.0f);
#pragma unroll
    for (int g = 0; g < 4; ++g) { Bx1[g][0] = z; Bx1[g][1] = z; Bh1[g] = z; }
  }
  if (wv < 2) {
#pragma unroll
    for (int g = 0; g < 4; ++g) {
      const int rb = g * H1 + 16 * wv;
#pragma unroll
      for (int kb = 0; kb < 2; ++kb) Bx1[g][kb] = loadB(Wih1, H0, rb, kb * 32, lane);
      Bh1[g] = loadB(Whh1, H1, rb, 0, lane);
      float b = bih1[rb + n] + bhh1[rb + n];
      hb1[g] = (g == 2) ? b : 0.5f * b;
    }
  }

  // ---- D-layout -> A-layout store mapping (constant per lane) ----
  const int kb_w = wv >> 1;
  int hit0, jt0; invA(16 * (wv & 1) + n, hit0, jt0);
  int hit1, jt1; invA(16 * wv + n, hit1, jt1);          // layer1 (wv<2)

  // ---- x staging: 16 rows x 8 float4 segs; each seg = 4 contiguous A-halves ----
  const int lrow = tid >> 3, lseg = tid & 7;
  int shi, sj; invA(lseg * 4, shi, sj);
  const int sIdx = (lrow + 16 * shi) * 16 + sj;         // 8B-aligned slot in xA
  const float* xrowp = x + (size_t)(rowbase + lrow) * Tsz * Fsz + lseg * 4;

  // ---- recurrent state in registers ----
  v8f c0, c1;
  float h0v[8], h1v[8];
#pragma unroll
  for (int e = 0; e < 8; ++e) { c0[e] = 0.f; c1[e] = 0.f; h0v[e] = 0.f; h1v[e] = 0.f; }

  const v8f zf = {0.f, 0.f, 0.f, 0.f, 0.f, 0.f, 0.f, 0.f};

#pragma unroll 1
  for (int t = 0; t < Tsz; ++t) {
    const int pb = t & 1;         // buffer written this step
    const int pr = pb ^ 1;        // state from t-1

    // 1) stage x_t tile -> A-layout bf16 LDS (coalesced load, 1 packed store)
    {
      float4 xv = *reinterpret_cast<const float4*>(xrowp + (size_t)t * Fsz);
      union { __bf16 b[4]; unsigned long long u; } pk;
      pk.b[0] = f2bf(xv.x); pk.b[1] = f2bf(xv.y);
      pk.b[2] = f2bf(xv.z); pk.b[3] = f2bf(xv.w);
      *reinterpret_cast<unsigned long long*>(&xA[pb][sIdx]) = pk.u;
      if (t + 1 < Tsz) __builtin_prefetch(xrowp + (size_t)(t + 1) * Fsz, 0, 0);
    }
    __syncthreads();   // [A] xA(t) ready; everyone done with t-1 buffers

    // 2) layer 0: gates = x_t @ Wih0^T + h0 @ Whh0^T   (this wave's 16 units)
    v16bf ax  = *reinterpret_cast<const v16bf*>(&xA[pb][lane * 16]);
    v16bf ah0 = *reinterpret_cast<const v16bf*>(&h0A[pr][0][lane * 16]);
    v16bf ah1 = *reinterpret_cast<const v16bf*>(&h0A[pr][1][lane * 16]);

    v8f acc[4];
#pragma unroll
    for (int g = 0; g < 4; ++g) acc[g] = wmma_bf16(ax,  Bx0[g],    zf);
#pragma unroll
    for (int g = 0; g < 4; ++g) acc[g] = wmma_bf16(ah0, Bh0[g][0], acc[g]);
#pragma unroll
    for (int g = 0; g < 4; ++g) acc[g] = wmma_bf16(ah1, Bh0[g][1], acc[g]);

#pragma unroll
    for (int e = 0; e < 8; ++e) {
      float iv = sigm_b(acc[0][e], hb0[0]);
      float fv = sigm_b(acc[1][e], hb0[1]);
      float gv = fast_tanh(acc[2][e] + hb0[2]);
      float ov = sigm_b(acc[3][e], hb0[3]);
      c0[e]  = fmaf(fv, c0[e], iv * gv);
      h0v[e] = ov * fast_tanh(c0[e]);
    }

    // 3) publish h0(t) into the OTHER buffer (no WAR with t-1 readers)
#pragma unroll
    for (int r = 0; r < 8; ++r) {
      int mr = r + 8 * hi;                                // D-layout row
      h0A[pb][kb_w][(mr + 16 * hit0) * 16 + jt0] = f2bf(h0v[r]);
    }
    __syncthreads();   // [C] h0A(t) visible

    // 4) layer 1 (waves 0,1): gates = h0_t @ Wih1^T + h1 @ Whh1^T
    if (wv < 2) {
      v16bf a0 = *reinterpret_cast<const v16bf*>(&h0A[pb][0][lane * 16]);
      v16bf a1 = *reinterpret_cast<const v16bf*>(&h0A[pb][1][lane * 16]);
      v16bf ah = *reinterpret_cast<const v16bf*>(&h1A[pr][lane * 16]);
      v8f acc1[4];
#pragma unroll
      for (int g = 0; g < 4; ++g) acc1[g] = wmma_bf16(a0, Bx1[g][0], zf);
#pragma unroll
      for (int g = 0; g < 4; ++g) acc1[g] = wmma_bf16(a1, Bx1[g][1], acc1[g]);
#pragma unroll
      for (int g = 0; g < 4; ++g) acc1[g] = wmma_bf16(ah, Bh1[g],    acc1[g]);

#pragma unroll
      for (int e = 0; e < 8; ++e) {
        float iv = sigm_b(acc1[0][e], hb1[0]);
        float fv = sigm_b(acc1[1][e], hb1[1]);
        float gv = fast_tanh(acc1[2][e] + hb1[2]);
        float ov = sigm_b(acc1[3][e], hb1[3]);
        c1[e]  = fmaf(fv, c1[e], iv * gv);
        h1v[e] = ov * fast_tanh(c1[e]);
      }
#pragma unroll
      for (int r = 0; r < 8; ++r) {
        int mr = r + 8 * hi;
        h1A[pb][(mr + 16 * hit1) * 16 + jt1] = f2bf(h1v[r]);
      }
    }
    // next iteration's [A] barrier orders h1A(t) writes vs t+1 readers
  }

  // ---- epilogue: (out[B,1], hT0[B,64], cT0[B,64], hT1[B,32], cT1[B,32]) ----
  float* outO   = dout;
  float* outHT0 = dout + Bsz;
  float* outCT0 = outHT0 + Bsz * H0;
  float* outHT1 = outCT0 + Bsz * H0;
  float* outCT1 = outHT1 + Bsz * H1;

#pragma unroll
  for (int r = 0; r < 8; ++r) {
    int row = rowbase + r + 8 * hi;
    int u   = 16 * wv + n;
    outHT0[(size_t)row * H0 + u] = h0v[r];
    outCT0[(size_t)row * H0 + u] = c0[r];
  }
  if (wv < 2) {
#pragma unroll
    for (int r = 0; r < 8; ++r) {
      int row = rowbase + r + 8 * hi;
      int u   = 16 * wv + n;
      outHT1[(size_t)row * H1 + u] = h1v[r];
      outCT1[(size_t)row * H1 + u] = c1[r];
      h1plain[(r + 8 * hi) * 32 + u] = h1v[r];
    }
  }
  __syncthreads();

  if (tid < 16) {
    float s = fcb[0];
#pragma unroll
    for (int k = 0; k < 32; ++k) s += h1plain[tid * 32 + k] * fcw[k];
    outO[rowbase + tid] = s;
  }
}

extern "C" void kernel_launch(void* const* d_in, const int* in_sizes, int n_in,
                              void* d_out, int out_size, void* d_ws, size_t ws_size,
                              hipStream_t stream) {
  (void)in_sizes; (void)n_in; (void)out_size; (void)d_ws; (void)ws_size;
  const float* x    = (const float*)d_in[0];
  const float* Wih0 = (const float*)d_in[1];
  const float* Whh0 = (const float*)d_in[2];
  const float* bih0 = (const float*)d_in[3];
  const float* bhh0 = (const float*)d_in[4];
  const float* Wih1 = (const float*)d_in[5];
  const float* Whh1 = (const float*)d_in[6];
  const float* bih1 = (const float*)d_in[7];
  const float* bhh1 = (const float*)d_in[8];
  const float* fcw  = (const float*)d_in[9];
  const float* fcb  = (const float*)d_in[10];

  dim3 grid(Bsz / 16), block(128);
  lstm2_fused_kernel<<<grid, block, 0, stream>>>(
      x, Wih0, Whh0, bih0, bhh0, Wih1, Whh1, bih1, bhh1, fcw, fcb, (float*)d_out);
}
